// sMAUCell_64201171140645
// MI455X (gfx1250) — matrix operand order repaired and checked by reading, not compiled, not run on hardware
//
#include <hip/hip_runtime.h>
#include <hip/hip_bf16.h>

#define Bn   16384
#define Hn   512
#define TAUn 5
#define EPSf 1e-5f

typedef __attribute__((ext_vector_type(16))) __bf16 v16bf;
typedef __attribute__((ext_vector_type(8)))  float  v8f;

union Frag {
    v16bf v;
    unsigned int u[8];
};

__device__ __forceinline__ unsigned short f32_to_bf16(float f) {
    union { float f; unsigned int u; } cv;
    cv.f = f;
    unsigned int u = cv.u;
    u += 0x7FFFu + ((u >> 16) & 1u);   // round-to-nearest-even
    return (unsigned short)(u >> 16);
}

__device__ __forceinline__ float sigmoidf(float x) {
    return 1.0f / (1.0f + __expf(-x));
}

// ---------------------------------------------------------------------------
// Pack fp32 -> bf16 (row-major copy)
// ---------------------------------------------------------------------------
__global__ __launch_bounds__(256) void pack_bf16_kernel(
    const float* __restrict__ src, unsigned short* __restrict__ dst, long n)
{
    long i = (long)blockIdx.x * 256 + threadIdx.x;
    long stride = (long)gridDim.x * 256;
    for (; i < n; i += stride) dst[i] = f32_to_bf16(src[i]);
}

// ---------------------------------------------------------------------------
// Pack fp32 W[K][N] -> bf16 Wt[N][K] (transpose so GEMM B-tiles are K-contiguous)
// ---------------------------------------------------------------------------
__global__ __launch_bounds__(256) void pack_transpose_bf16_kernel(
    const float* __restrict__ W, unsigned short* __restrict__ Wt, int K, int N)
{
    long total = (long)K * N;
    long stride = (long)gridDim.x * 256;
    for (long i = (long)blockIdx.x * 256 + threadIdx.x; i < total; i += stride) {
        long k = i / N;
        long n = i - k * N;
        Wt[n * (long)K + k] = f32_to_bf16(W[i]);
    }
}

// ---------------------------------------------------------------------------
// GEMM: C[M x N] = A[M x 512](bf16) @ Wt[N x 512](bf16, pre-transposed) + bias
// Block tile 128x128, 8 waves (4 in M x 2 in N), each wave 32x64 via 2x4 WMMA
// accumulators. K staged 32-deep through LDS.
// ---------------------------------------------------------------------------
__global__ __launch_bounds__(256) void gemm_bf16_kernel(
    const unsigned short* __restrict__ A,
    const unsigned short* __restrict__ Bt,
    const float* __restrict__ bias,
    float* __restrict__ C,
    int N)
{
    __shared__ uint4 As4[128 * 4];   // 128 rows x 32 bf16 = 8 KB
    __shared__ uint4 Bs4[128 * 4];   // 128 n-rows x 32 bf16 = 8 KB
    unsigned int* As = reinterpret_cast<unsigned int*>(As4);
    unsigned int* Bs = reinterpret_cast<unsigned int*>(Bs4);

    const int tid  = threadIdx.x;
    const int lane = tid & 31;
    const int wave = tid >> 5;
    const int wm   = wave & 3;   // M sub-tile: wm*32
    const int wn   = wave >> 2;  // N sub-tile: wn*64
    const int row0 = blockIdx.y * 128;
    const int col0 = blockIdx.x * 128;

    v8f acc[2][4];
    for (int mi = 0; mi < 2; ++mi)
        for (int ni = 0; ni < 4; ++ni)
            for (int e = 0; e < 8; ++e) acc[mi][ni][e] = 0.0f;

    const int mlane = lane & 15;
    // ISA 7.12.2: 16-bit A 16x32: lanes 0-15 hold K0..7,16..23; lanes 16-31 K8..15,24..31
    const int kbA = (lane < 16) ? 0 : 4;   // dword offset (K/2)
    // 16-bit B 32x16: same K order as A: lanes 0-15 hold K0..7,16..23; lanes 16-31 K8..15,24..31
    const int kbB = (lane < 16) ? 0 : 4;   // dword offset (K/2)

    for (int k0 = 0; k0 < 512; k0 += 32) {
        // ---- stage A tile (128 x 32 bf16) ----
        {
            int c = tid;
            for (int it = 0; it < 2; ++it, c += 256) {
                int r = c >> 2;
                int q = c & 3;
                const unsigned short* p = A + ((long)(row0 + r)) * 512 + k0 + q * 8;
                As4[r * 4 + q] = *reinterpret_cast<const uint4*>(p);
            }
        }
        // ---- stage B tile (128 x 32 bf16, n-major/K-contiguous) ----
        {
            int c = tid;
            for (int it = 0; it < 2; ++it, c += 256) {
                int r = c >> 2;
                int q = c & 3;
                const unsigned short* p = Bt + ((long)(col0 + r)) * 512 + k0 + q * 8;
                Bs4[r * 4 + q] = *reinterpret_cast<const uint4*>(p);
            }
        }
        // prefetch next K-slab into cache while we compute
        if (k0 + 32 < 512) {
            __builtin_prefetch(A + ((long)(row0 + (tid >> 1))) * 512 + k0 + 32, 0, 1);
            __builtin_prefetch(Bt + ((long)(col0 + (tid >> 1))) * 512 + k0 + 32, 0, 1);
        }
        __syncthreads();

        // ---- build fragments and issue WMMAs ----
        Frag afr[2], bfr[4];
        for (int mi = 0; mi < 2; ++mi) {
            const int m = wm * 32 + mi * 16 + mlane;
            const unsigned int* p = &As[m * 16 + kbA];
            afr[mi].u[0] = p[0]; afr[mi].u[1] = p[1];
            afr[mi].u[2] = p[2]; afr[mi].u[3] = p[3];
            afr[mi].u[4] = p[8]; afr[mi].u[5] = p[9];
            afr[mi].u[6] = p[10]; afr[mi].u[7] = p[11];
        }
        for (int ni = 0; ni < 4; ++ni) {
            const int n = wn * 64 + ni * 16 + mlane;
            const unsigned int* p = &Bs[n * 16 + kbB];
            for (int r = 0; r < 8; ++r) bfr[ni].u[r] = p[(r & 3) + ((r >> 2) << 3)];
        }
        for (int mi = 0; mi < 2; ++mi)
            for (int ni = 0; ni < 4; ++ni)
                acc[mi][ni] = __builtin_amdgcn_wmma_f32_16x16x32_bf16(
                    false, afr[mi].v, false, bfr[ni].v,
                    (short)0, acc[mi][ni], false, false);
        __syncthreads();
    }

    // ---- epilogue: + bias, fp32 out ----
    const int mbase = row0 + wm * 32;
    const int nbase = col0 + wn * 64;
    const int rsel  = (lane >> 4) * 8;   // lanes 16-31 hold M = 8..15
    for (int mi = 0; mi < 2; ++mi)
        for (int ni = 0; ni < 4; ++ni) {
            const int col = nbase + ni * 16 + mlane;
            const float bv = bias[col];
            for (int e = 0; e < 8; ++e) {
                const int row = mbase + mi * 16 + rsel + e;
                C[(long)row * N + col] = acc[mi][ni][e] + bv;
            }
        }
}

// ---------------------------------------------------------------------------
// Block-wide paired reduction (sum over 256 threads of two values)
// ---------------------------------------------------------------------------
__device__ __forceinline__ float2 block_reduce2(float a, float b,
                                                float* r1, float* r2, int tid)
{
    r1[tid] = a; r2[tid] = b;
    __syncthreads();
    for (int off = 128; off > 0; off >>= 1) {
        if (tid < off) { r1[tid] += r1[tid + off]; r2[tid] += r2[tid + off]; }
        __syncthreads();
    }
    float2 out = make_float2(r1[0], r2[0]);
    __syncthreads();
    return out;
}

// ---------------------------------------------------------------------------
// Fused: LN(s_pre)->s_next, LN(t_pre)->t_next, tau-attention softmax,
// T_trend, sigmoid fusion -> T_fusion (bf16 for next GEMM). 1 block per row.
// ---------------------------------------------------------------------------
__global__ __launch_bounds__(256) void ln_attn_fusion_kernel(
    const float* __restrict__ s_pre, const float* __restrict__ t_pre,
    const float* __restrict__ g_sn, const float* __restrict__ be_sn,
    const float* __restrict__ g_tn, const float* __restrict__ be_tn,
    const float* __restrict__ s_att, const float* __restrict__ t_att,
    const float* __restrict__ T_t,
    unsigned short* __restrict__ tf_bf)
{
    __shared__ float r1[256];
    __shared__ float r2[256];
    __shared__ float wsh[TAUn];

    const int b = blockIdx.x;
    const int tid = threadIdx.x;
    const int i0 = tid, i1 = tid + 256;
    const long base = (long)b * Hn;
    const float invH = 1.0f / (float)Hn;
    const float scale = 0.044194173824159216f;  // 1/sqrt(512)

    // ---- LN over s_pre ----
    float sp0 = s_pre[base + i0], sp1 = s_pre[base + i1];
    float2 st = block_reduce2(sp0 + sp1, sp0 * sp0 + sp1 * sp1, r1, r2, tid);
    float mu = st.x * invH;
    float inv = rsqrtf(st.y * invH - mu * mu + EPSf);
    const float sn0 = (sp0 - mu) * inv * g_sn[i0] + be_sn[i0];
    const float sn1 = (sp1 - mu) * inv * g_sn[i1] + be_sn[i1];

    // ---- LN over t_pre ----
    float tp0 = t_pre[base + i0], tp1 = t_pre[base + i1];
    float2 tt = block_reduce2(tp0 + tp1, tp0 * tp0 + tp1 * tp1, r1, r2, tid);
    mu = tt.x * invH;
    inv = rsqrtf(tt.y * invH - mu * mu + EPSf);
    const float tn0 = (tp0 - mu) * inv * g_tn[i0] + be_tn[i0];
    const float tn1 = (tp1 - mu) * inv * g_tn[i1] + be_tn[i1];

    // ---- attention logits: dot(s_att[t,b,:], s_next) * scale ----
    for (int t = 0; t < TAUn; ++t) {
        const long ab = ((long)t * Bn + b) * Hn;
        float p = s_att[ab + i0] * sn0 + s_att[ab + i1] * sn1;
        r1[tid] = p;
        __syncthreads();
        for (int off = 128; off > 0; off >>= 1) {
            if (tid < off) r1[tid] += r1[tid + off];
            __syncthreads();
        }
        if (tid == 0) wsh[t] = r1[0] * scale;
        __syncthreads();
    }

    // ---- softmax over tau (redundant per thread) ----
    float w[TAUn];
    float mx = wsh[0];
    for (int t = 1; t < TAUn; ++t) mx = fmaxf(mx, wsh[t]);
    float se = 0.0f;
    for (int t = 0; t < TAUn; ++t) { w[t] = __expf(wsh[t] - mx); se += w[t]; }
    const float rs = 1.0f / se;

    // ---- T_trend + sigmoid fusion ----
    float tr0 = 0.0f, tr1 = 0.0f;
    for (int t = 0; t < TAUn; ++t) {
        const long ab = ((long)t * Bn + b) * Hn;
        const float wt = w[t] * rs;
        tr0 += t_att[ab + i0] * wt;
        tr1 += t_att[ab + i1] * wt;
    }
    const float g0 = sigmoidf(tn0), g1 = sigmoidf(tn1);
    const float tf0 = T_t[base + i0] * g0 + (1.0f - g0) * tr0;
    const float tf1 = T_t[base + i1] * g1 + (1.0f - g1) * tr1;
    tf_bf[base + i0] = f32_to_bf16(tf0);
    tf_bf[base + i1] = f32_to_bf16(tf1);
}

// ---------------------------------------------------------------------------
// Fused: LN(T_concat_pre 1536), LN(S_concat_pre 1536), split/gates/residual,
// write T_new then S_new to d_out. 1 block per row; thread owns h, h+256 of
// each of the three 512-wide segments (6 elems per tensor).
// ---------------------------------------------------------------------------
__global__ __launch_bounds__(256) void ln_gate_out_kernel(
    const float* __restrict__ Tc, const float* __restrict__ Sc,
    const float* __restrict__ g_t, const float* __restrict__ be_t,
    const float* __restrict__ g_s, const float* __restrict__ be_s,
    const float* __restrict__ S_t, float* __restrict__ out)
{
    __shared__ float r1[256];
    __shared__ float r2[256];

    const int b = blockIdx.x;
    const int tid = threadIdx.x;
    const long base3 = (long)b * (3 * Hn);
    const long base  = (long)b * Hn;
    const float invN = 1.0f / (float)(3 * Hn);

    int cols[6];
    cols[0] = tid;             cols[1] = tid + 256;
    cols[2] = 512 + tid;       cols[3] = 512 + tid + 256;
    cols[4] = 1024 + tid;      cols[5] = 1024 + tid + 256;

    float tc[6], sc[6];
    float ts = 0.0f, tss = 0.0f, ss = 0.0f, sss = 0.0f;
    for (int j = 0; j < 6; ++j) {
        tc[j] = Tc[base3 + cols[j]];
        sc[j] = Sc[base3 + cols[j]];
        ts += tc[j]; tss += tc[j] * tc[j];
        ss += sc[j]; sss += sc[j] * sc[j];
    }
    float2 rt = block_reduce2(ts, tss, r1, r2, tid);
    float mu_t = rt.x * invN;
    float inv_t = rsqrtf(rt.y * invN - mu_t * mu_t + EPSf);
    float2 rsum = block_reduce2(ss, sss, r1, r2, tid);
    float mu_s = rsum.x * invN;
    float inv_s = rsqrtf(rsum.y * invN - mu_s * mu_s + EPSf);

    float tcn[6], scn[6];
    for (int j = 0; j < 6; ++j) {
        tcn[j] = (tc[j] - mu_t) * inv_t * g_t[cols[j]] + be_t[cols[j]];
        scn[j] = (sc[j] - mu_s) * inv_s * g_s[cols[j]] + be_s[cols[j]];
    }

    for (int p = 0; p < 2; ++p) {
        const int h = cols[p];                   // tid or tid+256
        const float tg = sigmoidf(tcn[p]);       // t_g
        const float sg = sigmoidf(scn[p]);       // s_g
        const float T_new = tg * tcn[2 + p] + (1.0f - tg) * scn[2 + p];
        const float S_new = sg * scn[4 + p] + (1.0f - sg) * tcn[4 + p]
                            + S_t[base + h];
        out[base + h] = T_new;
        out[(long)Bn * Hn + base + h] = S_new;
    }
}

// ---------------------------------------------------------------------------
// Host launch
// ---------------------------------------------------------------------------
extern "C" void kernel_launch(void* const* d_in, const int* in_sizes, int n_in,
                              void* d_out, int out_size, void* d_ws, size_t ws_size,
                              hipStream_t stream)
{
    (void)in_sizes; (void)n_in; (void)out_size; (void)ws_size;

    const float* T_t   = (const float*)d_in[0];
    const float* S_t   = (const float*)d_in[1];
    const float* t_att = (const float*)d_in[2];
    const float* s_att = (const float*)d_in[3];
    const float* W_tn  = (const float*)d_in[4];
    const float* b_tn  = (const float*)d_in[5];
    const float* g_tn  = (const float*)d_in[6];
    const float* be_tn = (const float*)d_in[7];
    const float* W_sn  = (const float*)d_in[8];
    const float* b_sn  = (const float*)d_in[9];
    const float* g_sn  = (const float*)d_in[10];
    const float* be_sn = (const float*)d_in[11];
    const float* W_t   = (const float*)d_in[12];
    const float* b_t   = (const float*)d_in[13];
    const float* g_t   = (const float*)d_in[14];
    const float* be_t  = (const float*)d_in[15];
    const float* W_s   = (const float*)d_in[16];
    const float* b_s   = (const float*)d_in[17];
    const float* g_s   = (const float*)d_in[18];
    const float* be_s  = (const float*)d_in[19];

    char* ws = (char*)d_ws;
    // --- workspace layout (bytes) ---
    const size_t OFF_WTN  = 0;                       // 512*512*2   = 0.5 MB
    const size_t OFF_WSN  = 524288;
    const size_t OFF_WT   = 1048576;                 // 512*1536*2  = 1.5 MB
    const size_t OFF_WS   = 2621440;
    const size_t OFF_TTB  = 4194304;                 // 16 MB bf16 T_t
    const size_t OFF_STB  = 20971520;                // 16 MB bf16 S_t
    const size_t OFF_TFB  = 37748736;                // 16 MB bf16 T_fusion
    const size_t OFF_SPRE = 54525952;                // 32 MB fp32 s_pre
    const size_t OFF_TPRE = 88080384;                // 32 MB fp32 t_pre
    const size_t OFF_TC   = 54525952;                // 96 MB (overlaps dead s/t_pre)
    const size_t OFF_SC   = 155189248;               // 96 MB

    unsigned short* wtn_t = (unsigned short*)(ws + OFF_WTN);
    unsigned short* wsn_t = (unsigned short*)(ws + OFF_WSN);
    unsigned short* wt_t  = (unsigned short*)(ws + OFF_WT);
    unsigned short* ws_t  = (unsigned short*)(ws + OFF_WS);
    unsigned short* tt_bf = (unsigned short*)(ws + OFF_TTB);
    unsigned short* st_bf = (unsigned short*)(ws + OFF_STB);
    unsigned short* tf_bf = (unsigned short*)(ws + OFF_TFB);
    float* s_pre = (float*)(ws + OFF_SPRE);
    float* t_pre = (float*)(ws + OFF_TPRE);
    float* Tc    = (float*)(ws + OFF_TC);
    float* Sc    = (float*)(ws + OFF_SC);

    // 1) pack weights (transposed) + activations to bf16
    pack_transpose_bf16_kernel<<<256, 256, 0, stream>>>(W_tn, wtn_t, Hn, Hn);
    pack_transpose_bf16_kernel<<<256, 256, 0, stream>>>(W_sn, wsn_t, Hn, Hn);
    pack_transpose_bf16_kernel<<<768, 256, 0, stream>>>(W_t,  wt_t,  Hn, 3 * Hn);
    pack_transpose_bf16_kernel<<<768, 256, 0, stream>>>(W_s,  ws_t,  Hn, 3 * Hn);
    pack_bf16_kernel<<<4096, 256, 0, stream>>>(T_t, tt_bf, (long)Bn * Hn);
    pack_bf16_kernel<<<4096, 256, 0, stream>>>(S_t, st_bf, (long)Bn * Hn);

    // 2) GEMM1: s_pre = S_t @ W_sn + b_sn ; t_pre = T_t @ W_tn + b_tn
    gemm_bf16_kernel<<<dim3(Hn / 128, Bn / 128), 256, 0, stream>>>(
        st_bf, wsn_t, b_sn, s_pre, Hn);
    gemm_bf16_kernel<<<dim3(Hn / 128, Bn / 128), 256, 0, stream>>>(
        tt_bf, wtn_t, b_tn, t_pre, Hn);

    // 3) fused LN + tau-attention + sigmoid fusion -> T_fusion (bf16)
    ln_attn_fusion_kernel<<<Bn, 256, 0, stream>>>(
        s_pre, t_pre, g_sn, be_sn, g_tn, be_tn, s_att, t_att, T_t, tf_bf);

    // 4) GEMM2: Tc = T_fusion @ W_t + b_t ; Sc = S_t @ W_s + b_s
    gemm_bf16_kernel<<<dim3(3 * Hn / 128, Bn / 128), 256, 0, stream>>>(
        tf_bf, wt_t, b_t, Tc, 3 * Hn);
    gemm_bf16_kernel<<<dim3(3 * Hn / 128, Bn / 128), 256, 0, stream>>>(
        st_bf, ws_t, b_s, Sc, 3 * Hn);

    // 5) fused dual-LN + gates + residual -> d_out (T_new, then S_new)
    ln_gate_out_kernel<<<Bn, 256, 0, stream>>>(
        Tc, Sc, g_t, be_t, g_s, be_s, S_t, (float*)d_out);
}
